// WeightDropLSTMCustom_46291157516574
// MI455X (gfx1250) — matrix operand order repaired and checked by reading, not compiled, not run on hardware
//
#include <hip/hip_runtime.h>

// ---------------------------------------------------------------------------
// WeightDrop(p=0) LSTM for MI455X (gfx1250, wave32, WMMA).
//   T=512, B=64, I=H=1024, 4H=4096.
// Strategy:
//   * pre-convert W_x, W_h, h0 (and optionally the whole input) to f16 in ws
//   * one fused kernel launch per timestep:
//       g = x_t @ Wx^T + h_{t-1} @ Wh^T + b   via v_wmma_f32_16x16x32_f16
//       gate nonlinearities + c/h update fused through LDS
//   * h state ping-pong (f16) to avoid intra-step RAW races across blocks
//   * outputs[t] written directly into d_out; final kernel fills h_f, c_f
// ---------------------------------------------------------------------------

#define T_STEPS 512
#define BATCH   64
#define ISIZE   1024
#define HSIZE   1024

typedef __attribute__((ext_vector_type(16))) _Float16 v16h;
typedef __attribute__((ext_vector_type(8)))  float    v8f;

union FragU {
  v16h     v;
  uint4    q[2];
  _Float16 h[16];
};

// A-operand (16x32 f16, MxK): lane holds row m=lane%16; K elems are two
// contiguous 8-chunks at k0 + 8*(lane>=16) and +16 beyond that.
__device__ __forceinline__ v16h load_a_h16(const _Float16* p) {
  FragU f;
  f.q[0] = *(const uint4*)(p);
  f.q[1] = *(const uint4*)(p + 16);
  return f.v;
}

// B-operand (32x16 f16, KxN): lane holds col n=lane%16; K elems are two
// contiguous 8-chunks at k0 + 8*(lane>=16) and +16 beyond that.
__device__ __forceinline__ v16h load_b_h16(const _Float16* p) {
  FragU f;
  f.q[0] = *(const uint4*)(p);
  f.q[1] = *(const uint4*)(p + 16);
  return f.v;
}

// A-operand loaded from fp32 memory with in-register cvt to f16.
__device__ __forceinline__ v16h load_a_f32(const float* p) {
  const float4 a = *(const float4*)(p);
  const float4 b = *(const float4*)(p + 4);
  const float4 c = *(const float4*)(p + 16);
  const float4 d = *(const float4*)(p + 20);
  FragU f;
  f.h[0]  = (_Float16)a.x;  f.h[1]  = (_Float16)a.y;
  f.h[2]  = (_Float16)a.z;  f.h[3]  = (_Float16)a.w;
  f.h[4]  = (_Float16)b.x;  f.h[5]  = (_Float16)b.y;
  f.h[6]  = (_Float16)b.z;  f.h[7]  = (_Float16)b.w;
  f.h[8]  = (_Float16)c.x;  f.h[9]  = (_Float16)c.y;
  f.h[10] = (_Float16)c.z;  f.h[11] = (_Float16)c.w;
  f.h[12] = (_Float16)d.x;  f.h[13] = (_Float16)d.y;
  f.h[14] = (_Float16)d.z;  f.h[15] = (_Float16)d.w;
  return f.v;
}

#define WMMA_F16(A, B, C) \
  __builtin_amdgcn_wmma_f32_16x16x32_f16(false, (A), false, (B), (short)0, (C), false, false)

__device__ __forceinline__ float fsigmoid(float x) { return 1.0f / (1.0f + __expf(-x)); }
__device__ __forceinline__ float ftanh(float x) {
  float ax = fabsf(x);
  float e  = __expf(-2.0f * ax);
  float r  = (1.0f - e) / (1.0f + e);
  return copysignf(r, x);
}

// ---------------------------------------------------------------------------
// Fused per-timestep kernel. Grid: 64 blocks x 256 threads (8 waves).
// Block `bid` owns h-columns [bid*16, bid*16+16) for ALL four gates.
// Wave w: gate = w>>1, mhalf = w&1 -> two 16x16 C tiles (rows mhalf*32..+31,
// cols gate*1024 + bid*16 .. +16).
// ---------------------------------------------------------------------------
__global__ __launch_bounds__(256)
void lstm_step_kernel(const float* __restrict__ x_all,        // [T,B,I] fp32
                      const _Float16* __restrict__ x_h16,     // optional [T,B,I] f16 (may be null)
                      const _Float16* __restrict__ Wx,        // [4096,1024] f16, K-contig rows
                      const _Float16* __restrict__ Wh,        // [4096,1024] f16
                      const float* __restrict__ b_x,          // [4096]
                      const float* __restrict__ b_h,          // [4096]
                      const _Float16* __restrict__ h_in,      // [B,H] f16
                      _Float16* __restrict__ h_out,           // [B,H] f16
                      float* __restrict__ c_state,            // [B,H] fp32 (in/out)
                      float* __restrict__ out,                // [T,B,H] fp32
                      int t)
{
  __shared__ float gsm[64][64];   // [batch row][gate*16 + col_local]

  const int tid   = threadIdx.x;
  const int wave  = tid >> 5;
  const int lane  = tid & 31;
  const int ln16  = lane & 15;
  const int hi    = lane >> 4;          // 0 | 1
  const int gate  = wave >> 1;          // 0..3  (i,f,o,n)
  const int mhalf = wave & 1;           // 0 | 1
  const int bid   = blockIdx.x;         // 0..63

  const int gcol = gate * 1024 + bid * 16 + ln16;   // global column in [0,4096)

  const _Float16* wxr = Wx + (size_t)gcol * ISIZE;
  const _Float16* whr = Wh + (size_t)gcol * HSIZE;

  const int m0 = mhalf * 32 + ln16;     // A row for tile 0 (this lane)
  const int m1 = m0 + 16;               // A row for tile 1
  const int akc = hi * 8;               // A per-lane K sub-offset
  const int bkc = hi * 8;               // B per-lane K sub-offset

  v8f acc0 = {0.f, 0.f, 0.f, 0.f, 0.f, 0.f, 0.f, 0.f};
  v8f acc1 = {0.f, 0.f, 0.f, 0.f, 0.f, 0.f, 0.f, 0.f};

  // ---- phase 1: x_t @ Wx^T ------------------------------------------------
  if (x_h16) {
    const _Float16* xr0 = x_h16 + ((size_t)t * BATCH + m0) * ISIZE;
    const _Float16* xr1 = x_h16 + ((size_t)t * BATCH + m1) * ISIZE;
    for (int k0 = 0; k0 < ISIZE; k0 += 32) {
      __builtin_prefetch(wxr + k0 + 256, 0, 3);
      v16h B  = load_b_h16(wxr + k0 + bkc);
      v16h A0 = load_a_h16(xr0 + k0 + akc);
      v16h A1 = load_a_h16(xr1 + k0 + akc);
      acc0 = WMMA_F16(A0, B, acc0);
      acc1 = WMMA_F16(A1, B, acc1);
    }
  } else {
    const float* xr0 = x_all + ((size_t)t * BATCH + m0) * ISIZE;
    const float* xr1 = x_all + ((size_t)t * BATCH + m1) * ISIZE;
    for (int k0 = 0; k0 < ISIZE; k0 += 32) {
      __builtin_prefetch(wxr + k0 + 256, 0, 3);
      v16h B  = load_b_h16(wxr + k0 + bkc);
      v16h A0 = load_a_f32(xr0 + k0 + akc);
      v16h A1 = load_a_f32(xr1 + k0 + akc);
      acc0 = WMMA_F16(A0, B, acc0);
      acc1 = WMMA_F16(A1, B, acc1);
    }
  }

  // ---- phase 2: h_{t-1} @ Wh^T -------------------------------------------
  {
    const _Float16* hr0 = h_in + (size_t)m0 * HSIZE;
    const _Float16* hr1 = h_in + (size_t)m1 * HSIZE;
    for (int k0 = 0; k0 < HSIZE; k0 += 32) {
      __builtin_prefetch(whr + k0 + 256, 0, 3);
      v16h B  = load_b_h16(whr + k0 + bkc);
      v16h A0 = load_a_h16(hr0 + k0 + akc);
      v16h A1 = load_a_h16(hr1 + k0 + akc);
      acc0 = WMMA_F16(A0, B, acc0);
      acc1 = WMMA_F16(A1, B, acc1);
    }
  }

  // ---- stage pre-activations (+bias) into LDS ----------------------------
  // C/D layout: lane n = lane%16, rows m = 8*(lane>=16) + r  (r = VGPR idx)
  const float bias = b_x[gcol] + b_h[gcol];
  const int mb = hi * 8;
#pragma unroll
  for (int r = 0; r < 8; ++r) {
    gsm[mhalf * 32 +      mb + r][gate * 16 + ln16] = acc0[r] + bias;
    gsm[mhalf * 32 + 16 + mb + r][gate * 16 + ln16] = acc1[r] + bias;
  }
  __syncthreads();

  // ---- fused gate nonlinearities + state update --------------------------
  for (int e = tid; e < BATCH * 16; e += 256) {
    const int b = e >> 4;
    const int j = e & 15;
    const float xi = gsm[b][j];
    const float xf = gsm[b][16 + j];
    const float xo = gsm[b][32 + j];
    const float xn = gsm[b][48 + j];
    const float ig = fsigmoid(xi);
    const float fg = fsigmoid(xf);
    const float og = fsigmoid(xo);
    const float ng = ftanh(xn);
    const int hcol = bid * 16 + j;
    const int idx  = b * HSIZE + hcol;
    const float cn = fg * c_state[idx] + ig * ng;
    const float hn = og * ftanh(cn);
    c_state[idx] = cn;
    out[((size_t)t * BATCH + b) * HSIZE + hcol] = hn;
    h_out[idx] = (_Float16)hn;
  }
}

// ---------------------------------------------------------------------------
// Pre-/post-pass helpers
// ---------------------------------------------------------------------------
__global__ void cvt_f32_to_f16_kernel(const float* __restrict__ src,
                                      _Float16* __restrict__ dst, int n) {
  int i = blockIdx.x * blockDim.x + threadIdx.x;
  if (i < n) dst[i] = (_Float16)src[i];
}

__global__ void copy_f32_kernel(const float* __restrict__ src,
                                float* __restrict__ dst, int n) {
  int i = blockIdx.x * blockDim.x + threadIdx.x;
  if (i < n) dst[i] = src[i];
}

__global__ void finalize_kernel(const float* __restrict__ h_last,
                                const float* __restrict__ c_state,
                                float* __restrict__ h_f,
                                float* __restrict__ c_f, int n) {
  int i = blockIdx.x * blockDim.x + threadIdx.x;
  if (i < n) {
    h_f[i] = h_last[i];
    c_f[i] = c_state[i];
  }
}

// ---------------------------------------------------------------------------
// Host launcher
// ---------------------------------------------------------------------------
extern "C" void kernel_launch(void* const* d_in, const int* in_sizes, int n_in,
                              void* d_out, int out_size, void* d_ws, size_t ws_size,
                              hipStream_t stream) {
  (void)in_sizes; (void)n_in; (void)out_size;

  const float* x   = (const float*)d_in[0];   // [512,64,1024]
  const float* h0  = (const float*)d_in[1];   // [64,1024]
  const float* c0  = (const float*)d_in[2];   // [64,1024]
  const float* Wx  = (const float*)d_in[3];   // [4096,1024]
  const float* Wh  = (const float*)d_in[4];   // [4096,1024]
  const float* bx  = (const float*)d_in[5];   // [4096]
  const float* bh  = (const float*)d_in[6];   // [4096]
  float* out = (float*)d_out;                 // outputs | h_f | c_f

  char* ws = (char*)d_ws;
  // ws layout (bytes):
  //   [0,        8388608)   Wx f16
  //   [8388608,  16777216)  Wh f16
  //   [16777216, 16908288)  h ping (f16)
  //   [16908288, 17039360)  h pong (f16)
  //   [17039360, 17301504)  c state (fp32)
  //   [17301504, 84410368)  optional x in f16 (only if ws is big enough)
  _Float16* Wx_h = (_Float16*)(ws);
  _Float16* Wh_h = (_Float16*)(ws + 8388608);
  _Float16* hb0  = (_Float16*)(ws + 16777216);
  _Float16* hb1  = (_Float16*)(ws + 16908288);
  float*    cst  = (float*)   (ws + 17039360);
  _Float16* x_h  = nullptr;
  const size_t need_xh = 17301504ull + (size_t)T_STEPS * BATCH * ISIZE * 2;
  if (ws_size >= need_xh) x_h = (_Float16*)(ws + 17301504);

  const int WN = 4096 * 1024;        // weight element count
  cvt_f32_to_f16_kernel<<<(WN + 255) / 256, 256, 0, stream>>>(Wx, Wx_h, WN);
  cvt_f32_to_f16_kernel<<<(WN + 255) / 256, 256, 0, stream>>>(Wh, Wh_h, WN);
  cvt_f32_to_f16_kernel<<<(BATCH * HSIZE + 255) / 256, 256, 0, stream>>>(h0, hb0, BATCH * HSIZE);
  copy_f32_kernel<<<(BATCH * HSIZE + 255) / 256, 256, 0, stream>>>(c0, cst, BATCH * HSIZE);
  if (x_h) {
    const int XN = T_STEPS * BATCH * ISIZE;
    cvt_f32_to_f16_kernel<<<(XN + 255) / 256, 256, 0, stream>>>(x, x_h, XN);
  }

  for (int t = 0; t < T_STEPS; ++t) {
    _Float16* hin  = (t & 1) ? hb1 : hb0;
    _Float16* hout = (t & 1) ? hb0 : hb1;
    lstm_step_kernel<<<64, 256, 0, stream>>>(x, x_h, Wx_h, Wh_h, bx, bh,
                                             hin, hout, cst, out, t);
  }

  const size_t outs = (size_t)T_STEPS * BATCH * HSIZE;   // 33554432
  finalize_kernel<<<(BATCH * HSIZE + 255) / 256, 256, 0, stream>>>(
      out + (size_t)(T_STEPS - 1) * BATCH * HSIZE,       // h_last = outputs[511]
      cst,
      out + outs,                                        // h_f
      out + outs + BATCH * HSIZE,                        // c_f
      BATCH * HSIZE);
}
